// DiffJPEG_53841710023022
// MI455X (gfx1250) — matrix operand order repaired and checked by reading, not compiled, not run on hardware
//
#include <hip/hip_runtime.h>
#include <math.h>

typedef __attribute__((ext_vector_type(16))) _Float16 v16h;
typedef __attribute__((ext_vector_type(8)))  float    v8f;

// JPEG luma quant table (row-major as in the python source, before the .T)
__constant__ float c_ytab[64] = {
  16,11,10,16,24,40,51,61,
  12,12,14,19,26,58,60,55,
  14,13,16,24,40,57,69,56,
  14,17,22,29,51,87,80,62,
  18,22,37,56,68,109,103,77,
  24,35,55,64,81,104,113,92,
  49,64,78,87,103,121,120,101,
  72,92,95,98,112,100,103,99};
__constant__ float c_ctab4[16] = {
  17,18,24,47, 18,21,26,66, 24,26,56,99, 47,66,99,99};

union Row8    { float4 q[2]; float f[8]; };
union HalfPk  { _Float16 h[2]; unsigned u; };
union Frag16  { unsigned u[8]; float4 q[2]; v16h v; };

// Load one f16 A-fragment (16x32) for V_WMMA_F32_16X16X32_F16 from an LDS
// row-major [64][64] matrix. Lane (l,H): elems 0..15 = M[mt*16+l][kt*32+16H+0..15]
// (contiguous k order, matching the B fragments).
__device__ __forceinline__ v16h load_frag_A(const _Float16* Msrc, int mt, int kt,
                                            int l, int H) {
  Frag16 u;
  const _Float16* p = Msrc + ((mt * 16 + l) * 64 + kt * 32 + 16 * H);
  u.q[0] = *(const float4*)p;
  u.q[1] = *(const float4*)(p + 8);
  return u.v;
}

__global__ __launch_bounds__(256, 4)
void diffjpeg_fused(const float* __restrict__ img, float* __restrict__ out,
                    int nGroups) {
  // LDS: D (f16 [k][p]), D^T (f16 [p][k]), quant + reciprocal tables.
  __shared__ __align__(16) _Float16 sDh[4096];
  __shared__ __align__(16) _Float16 sDt[4096];
  __shared__ __align__(16) float    sQt[192];
  __shared__ __align__(16) float    sRc[192];

  // ---- build constant tables once per workgroup ----
  const float PI = 3.14159265358979323846f;
  for (int i = threadIdx.x; i < 4096; i += 256) {
    int k = i >> 6, p = i & 63;
    int u = k >> 3, v = k & 7;
    int x = p >> 3, y = p & 7;
    float au = (u == 0) ? 0.70710678118654752f : 1.0f;
    float av = (v == 0) ? 0.70710678118654752f : 1.0f;
    float val = au * av * 0.25f *
                cosf((2 * x + 1) * u * PI / 16.0f) *
                cosf((2 * y + 1) * v * PI / 16.0f);
    sDh[k * 64 + p] = (_Float16)val;   // A1 = D
    sDt[p * 64 + k] = (_Float16)val;   // A2 = D^T
  }
  if (threadIdx.x < 192) {
    int c = threadIdx.x >> 6, k = threadIdx.x & 63;
    int u = k >> 3, v = k & 7;
    float q = (c == 0) ? c_ytab[v * 8 + u]                      // .T applied
                       : ((u < 4 && v < 4) ? c_ctab4[v * 4 + u] : 99.0f);
    sQt[threadIdx.x] = q;
    sRc[threadIdx.x] = 1.0f / q;
  }
  __syncthreads();

  const int lane = threadIdx.x & 31;
  const int l    = lane & 15;       // block index within group / N-lane
  const int H    = lane >> 4;       // lane half
  const int g    = blockIdx.x * 8 + (threadIdx.x >> 5);   // one group per wave
  if (g >= nGroups) return;         // wave-uniform

  const size_t plane = (size_t)512 * 512;
  const int b   = g >> 8;           // batch
  const int t   = g & 255;
  const int bh  = t >> 2;           // block row 0..63
  const int bw0 = (t & 3) << 4;     // first of 16 block cols
  const size_t base = (size_t)(b * 3) * plane + (size_t)(bh * 8) * 512
                    + (size_t)((bw0 + l) * 8);

  // ---- phase 1: load RGB, color-convert, build B1 = flat^T fragments ----
  // B1 frag kt, lane (l,H): elems e=0..15 = flat[block l][p], p=kt*32+16H+e
  v16h B1[3][2];
#pragma unroll
  for (int kt = 0; kt < 2; ++kt) {
    const int x0 = kt * 4 + 2 * H;      // first of 2 block rows this lane owns
    Row8 cr_[2], cg_[2], cb_[2];
#pragma unroll
    for (int rr = 0; rr < 2; ++rr) {
      const float* p0 = img + base + (size_t)(x0 + rr) * 512;
      cr_[rr].q[0] = *(const float4*)(p0);
      cr_[rr].q[1] = *(const float4*)(p0 + 4);
      cg_[rr].q[0] = *(const float4*)(p0 + plane);
      cg_[rr].q[1] = *(const float4*)(p0 + plane + 4);
      cb_[rr].q[0] = *(const float4*)(p0 + 2 * plane);
      cb_[rr].q[1] = *(const float4*)(p0 + 2 * plane + 4);
    }
#pragma unroll
    for (int e = 0; e < 16; ++e) {
      const int rr = e >> 3, cc = e & 7;
      float R  = fminf(fmaxf(cr_[rr].f[cc], 0.f), 1.f);
      float G  = fminf(fmaxf(cg_[rr].f[cc], 0.f), 1.f);
      float Bv = fminf(fmaxf(cb_[rr].f[cc], 0.f), 1.f);
      float y  = 0.299f * R + 0.587f * G + 0.114f * Bv;
      // fold (*255 - 128) into the conversion
      B1[0][kt][e] = (_Float16)(255.0f * y - 128.0f);
      B1[1][kt][e] = (_Float16)(143.820f * (Bv - y) - 0.5f);   // 0.564*255
      B1[2][kt][e] = (_Float16)(181.815f * (R  - y) - 0.5f);   // 0.713*255
    }
  }
  __builtin_amdgcn_sched_barrier(0);

  // ---- phase 2: GEMM1  dct^T = D @ flat^T, fused quant/round/dequant ----
  // Per channel, per coeff tile: accumulate (8 regs), quantize immediately,
  // die into packed-f16 P. Only one f32 accumulator tile live at a time.
  unsigned P[3][4][4];
  {
    v16h A1[4][2];
#pragma unroll
    for (int mt = 0; mt < 4; ++mt)
#pragma unroll
      for (int kt = 0; kt < 2; ++kt)
        A1[mt][kt] = load_frag_A(sDh, mt, kt, l, H);

#pragma unroll
    for (int ch = 0; ch < 3; ++ch) {
#pragma unroll
      for (int mt = 0; mt < 4; ++mt) {
        v8f acc = (v8f){0,0,0,0,0,0,0,0};
#pragma unroll
        for (int kt = 0; kt < 2; ++kt)
          acc = __builtin_amdgcn_wmma_f32_16x16x32_f16(
              false, A1[mt][kt], false, B1[ch][kt], (short)0, acc, false, false);
        // acc reg r holds coeff (mt*16 + 8H + r) of block l
        Row8 qv, rv;
        const float* qp = sQt + ch * 64 + mt * 16 + 8 * H;
        const float* rp = sRc + ch * 64 + mt * 16 + 8 * H;
        qv.q[0] = *(const float4*)qp;  qv.q[1] = *(const float4*)(qp + 4);
        rv.q[0] = *(const float4*)rp;  rv.q[1] = *(const float4*)(rp + 4);
#pragma unroll
        for (int j = 0; j < 4; ++j) {
          HalfPk pk;
          pk.h[0] = (_Float16)(rintf(acc[2*j]   * rv.f[2*j])   * qv.f[2*j]);
          pk.h[1] = (_Float16)(rintf(acc[2*j+1] * rv.f[2*j+1]) * qv.f[2*j+1]);
          P[ch][mt][j] = pk.u;
        }
      }
    }
  }
  __builtin_amdgcn_sched_barrier(0);

  // ---- phase 3: C-layout -> B-layout via half-wave swap (no LDS) ----
  v16h B2[3][2];
#pragma unroll
  for (int ch = 0; ch < 3; ++ch) {
    unsigned S[4][4];
#pragma unroll
    for (int mt = 0; mt < 4; ++mt)
#pragma unroll
      for (int j = 0; j < 4; ++j)
        S[mt][j] = (unsigned)__shfl_xor((int)P[ch][mt][j], 16, 32);
#pragma unroll
    for (int kt = 0; kt < 2; ++kt) {
      Frag16 w;
#pragma unroll
      for (int j = 0; j < 4; ++j) {
        w.u[j]     = (H == 0) ? P[ch][2*kt][j]     : S[2*kt + 1][j];
        w.u[4 + j] = (H == 0) ? S[2*kt][j]         : P[ch][2*kt + 1][j];
      }
      B2[ch][kt] = w.v;
    }
  }
  __builtin_amdgcn_sched_barrier(0);

  // ---- phase 4: GEMM2  out^T = D^T @ deq^T ----
  v8f o[3][4];
#pragma unroll
  for (int ch = 0; ch < 3; ++ch)
#pragma unroll
    for (int pt = 0; pt < 4; ++pt)
      o[ch][pt] = (v8f){0,0,0,0,0,0,0,0};
  {
    v16h A2[4][2];
#pragma unroll
    for (int pt = 0; pt < 4; ++pt)
#pragma unroll
      for (int kt = 0; kt < 2; ++kt)
        A2[pt][kt] = load_frag_A(sDt, pt, kt, l, H);
#pragma unroll
    for (int pt = 0; pt < 4; ++pt)
#pragma unroll
      for (int kt = 0; kt < 2; ++kt)
#pragma unroll
        for (int ch = 0; ch < 3; ++ch)
          o[ch][pt] = __builtin_amdgcn_wmma_f32_16x16x32_f16(
              false, A2[pt][kt], false, B2[ch][kt], (short)0, o[ch][pt], false, false);
  }
  __builtin_amdgcn_sched_barrier(0);

  // ---- phase 5: YCbCr -> RGB, clamp, coalesced stores ----
  // o[ch][pt] reg r holds pixel p = pt*16 + 8H + r of block l
  // -> block row x = 2*pt + H, col y = r  (one full row per (pt) tile)
#pragma unroll
  for (int pt = 0; pt < 4; ++pt) {
    const int x = 2 * pt + H;
    Row8 orow[3];
#pragma unroll
    for (int r = 0; r < 8; ++r) {
      float yy = o[0][pt][r] + 128.0f;
      float cb = o[1][pt][r] + 0.5f;
      float cr = o[2][pt][r] + 0.5f;
      float Rv = (yy + 1.403f * cr) * (1.0f / 255.0f);
      float Gv = (yy - 0.714f * cr - 0.344f * cb) * (1.0f / 255.0f);
      float Bv = (yy + 1.773f * cb) * (1.0f / 255.0f);
      orow[0].f[r] = fminf(fmaxf(Rv, 0.f), 1.f);
      orow[1].f[r] = fminf(fmaxf(Gv, 0.f), 1.f);
      orow[2].f[r] = fminf(fmaxf(Bv, 0.f), 1.f);
    }
    float* po = out + base + (size_t)x * 512;
    *(float4*)(po)                 = orow[0].q[0];
    *(float4*)(po + 4)             = orow[0].q[1];
    *(float4*)(po + plane)         = orow[1].q[0];
    *(float4*)(po + plane + 4)     = orow[1].q[1];
    *(float4*)(po + 2 * plane)     = orow[2].q[0];
    *(float4*)(po + 2 * plane + 4) = orow[2].q[1];
  }
}

extern "C" void kernel_launch(void* const* d_in, const int* in_sizes, int n_in,
                              void* d_out, int out_size, void* d_ws, size_t ws_size,
                              hipStream_t stream) {
  (void)in_sizes; (void)n_in; (void)d_ws; (void)ws_size; (void)out_size;
  const float* img = (const float*)d_in[0];
  float* out = (float*)d_out;
  // 32 batches * 64 block-rows * 64 block-cols / 16 blocks-per-wave-group
  const int nGroups = 32 * 64 * 64 / 16;          // 8192
  const int blocks  = nGroups / 8;                 // 8 wave32 per block -> 1024
  hipLaunchKernelGGL(diffjpeg_fused, dim3(blocks), dim3(256), 0, stream,
                     img, out, nGroups);
}
